// MLAInference_64132451664700
// MI455X (gfx1250) — compile-verified, matrix operand order checked by reading, not run on hardware
//
#include <hip/hip_runtime.h>
#include <hip/hip_bf16.h>

// ---------------- problem constants ----------------
#define SEQ   2048
#define DMODEL 2048
#define NH    16
#define DHEAD 128
#define DKV   512
#define DQC   1536
#define DROT  64
#define HC    (NH * DKV)          // 8192, concatenated head-latent dim

typedef __attribute__((ext_vector_type(16))) __bf16 bf16x16;
typedef __attribute__((ext_vector_type(8)))  float  floatx8;
typedef unsigned int uint32x4 __attribute__((ext_vector_type(4)));
typedef int          int32x8  __attribute__((ext_vector_type(8)));
typedef int          int32x4  __attribute__((ext_vector_type(4)));

// ---------------- WMMA helpers ----------------
__device__ inline floatx8 wmma_bf16(bf16x16 a, bf16x16 b, floatx8 c) {
  return __builtin_amdgcn_wmma_f32_16x16x32_bf16(false, a, false, b,
                                                 (short)0, c, false, false);
}

// K-pair base index for 16-bit A/B fragment layout (ISA 7.12.2)
__device__ inline int wmma_kbase(int j, int half) {
  return (j < 4) ? (8 * half + 2 * j) : (16 + 8 * half + 2 * (j - 4));
}

// Load a 16x32 fragment from a row-major [row][k] tile (LDS or global).
__device__ inline bf16x16 load_frag_rm(const __bf16* base, long ld) {
  int lane = threadIdx.x & 31;
  int half = lane >> 4;
  int r    = lane & 15;
  bf16x16 f;
#pragma unroll
  for (int j = 0; j < 8; ++j) {
    int k = wmma_kbase(j, half);
    f[2 * j]     = base[(long)r * ld + k];
    f[2 * j + 1] = base[(long)r * ld + k + 1];
  }
  return f;
}

// Load a 32x16 B-fragment from a k-major tile: element (k,n) at base[k*ld+n].
__device__ inline bf16x16 load_frag_km(const __bf16* base, long ld) {
  int lane = threadIdx.x & 31;
  int half = lane >> 4;
  int n    = lane & 15;
  bf16x16 f;
#pragma unroll
  for (int j = 0; j < 8; ++j) {
    int k = wmma_kbase(j, half);
    f[2 * j]     = base[(long)k * ld + n];
    f[2 * j + 1] = base[(long)(k + 1) * ld + n];
  }
  return f;
}

__device__ inline floatx8 zero8() {
  floatx8 z;
#pragma unroll
  for (int j = 0; j < 8; ++j) z[j] = 0.0f;
  return z;
}

// ---------------- CDNA5 async / tensor data movement ----------------
// Async global->LDS copy (no VGPR data), tracked by ASYNCcnt.
__device__ inline void async_load_b128(unsigned lds_addr, const void* gaddr) {
  asm volatile("global_load_async_to_lds_b128 %0, %1, off"
               :: "v"(lds_addr), "v"(gaddr) : "memory");
}
__device__ inline void wait_async0() {
  asm volatile("s_wait_asynccnt 0x0" ::: "memory");
}
__device__ inline unsigned lds_addr_of(const void* p) {
  // generic LDS pointer: low 32 bits are the LDS byte offset
  return (unsigned)(unsigned long long)p;
}

// Tensor Data Mover: 2D tile [rows x 512 bf16] from global into LDS with
// 8-element row padding (LDS row stride 520 bf16 = 1040 B).
// pad_interval=7 (256 dwords = one 1024B row), pad_amount=3 (4 dwords = 8 bf16)
__device__ inline void tdm_load_kv_tile(unsigned lds_byte_addr,
                                        const __bf16* gsrc, int rows) {
  unsigned long long ga = (unsigned long long)gsrc;
  uint32x4 g0;
  g0[0] = 1u;                                   // count=1 (valid descriptor)
  g0[1] = lds_byte_addr;                        // lds_addr
  g0[2] = (unsigned)(ga & 0xffffffffu);         // global_addr[31:0]
  g0[3] = (unsigned)((ga >> 32) & 0x01ffffffu)  // global_addr[56:32]
          | 0x80000000u;                        // type=2 ("image")
  int32x8 g1;
  g1[0] = (int)0x07D10000u;  // data_size=2B, pad_enable, interval=7, amount=3
  g1[1] = (int)((DKV & 0xffff) << 16);          // tensor_dim0 = 512
  g1[2] = (int)((SEQ & 0xffff) << 16);          // tensor_dim1 = 2048
  g1[3] = (int)((DKV & 0xffff) << 16);          // tile_dim0 = 512
  g1[4] = rows & 0xffff;                        // tile_dim1 = rows
  g1[5] = DKV;                                  // tensor_dim0_stride = 512
  g1[6] = 0;
  g1[7] = 0;
  int32x4 gz;
  gz[0] = 0; gz[1] = 0; gz[2] = 0; gz[3] = 0;   // 2D: groups 2/3 unused
  int32x8 gz8;
#pragma unroll
  for (int i = 0; i < 8; ++i) gz8[i] = 0;
  // 6-arg variant on this toolchain: (g0, g1, g2, g3, g4, cpol)
  __builtin_amdgcn_tensor_load_to_lds(g0, g1, gz, gz, gz8, 0);
}

// ---------------- f32 -> bf16 convert / transpose ----------------
// out[((i0*d1)+i1)*d2 + i2] = in[i0*s0 + i1*s1 + i2*s2]
__global__ __launch_bounds__(256)
void conv_bf16_3d(const float* __restrict__ in, __bf16* __restrict__ out,
                  int d1, int d2, long s0, long s1, long s2) {
  long i0 = blockIdx.x / d1;
  long i1 = blockIdx.x % d1;
  const float* src = in + i0 * s0 + i1 * s1;
  __bf16* dst = out + (i0 * (long)d1 + i1) * (long)d2;
  for (int i2 = threadIdx.x; i2 < d2; i2 += 256)
    dst[i2] = (__bf16)src[(long)i2 * s2];
}

// ---------------- generic bf16 WMMA GEMM ----------------
// C[m,n] = alpha * sum_k A[m,k] * B[n,k]   (unit K-stride operands)
#define GM_BM 128
#define GM_BN 128
#define GM_BK 32

__global__ __launch_bounds__(256)
void gemm_bf16_wmma(const __bf16* __restrict__ A, long bsA, int lda,
                    const __bf16* __restrict__ B, long bsB, int ldb,
                    void* __restrict__ Cv, long bsC, int ldc,
                    int M, int N, int K, int c_bf16, float alpha) {
  __shared__ __bf16 As[GM_BM][GM_BK];
  __shared__ __bf16 Bs[GM_BN][GM_BK];

  A += (long)blockIdx.z * bsA;
  B += (long)blockIdx.z * bsB;
  const long coff = (long)blockIdx.z * bsC;

  const int m0 = blockIdx.y * GM_BM;
  const int n0 = blockIdx.x * GM_BN;
  const int tid  = threadIdx.x;
  const int wave = tid >> 5;
  const int wr   = wave >> 1;  // 0..3 -> M offset 32*wr
  const int wc   = wave & 1;   // 0..1 -> N offset 64*wc
  const int lane = tid & 31;
  const int half = lane >> 4;
  const int l15  = lane & 15;

  floatx8 acc[2][4];
#pragma unroll
  for (int i = 0; i < 2; ++i)
#pragma unroll
    for (int j = 0; j < 4; ++j) acc[i][j] = zero8();

  for (int k0 = 0; k0 < K; k0 += GM_BK) {
    // async-stage A/B tiles: 512 chunks of 8 bf16 each, 2 chunks/thread/matrix
#pragma unroll
    for (int c = 0; c < 2; ++c) {
      int idx = tid + c * 256;      // 0..511
      int r   = idx >> 2;
      int kb  = (idx & 3) * 8;
      if (m0 + r < M)
        async_load_b128(lds_addr_of(&As[r][kb]),
                        A + (long)(m0 + r) * lda + (k0 + kb));
      else
        *(uint4*)&As[r][kb] = make_uint4(0, 0, 0, 0);
      if (n0 + r < N)
        async_load_b128(lds_addr_of(&Bs[r][kb]),
                        B + (long)(n0 + r) * ldb + (k0 + kb));
      else
        *(uint4*)&Bs[r][kb] = make_uint4(0, 0, 0, 0);
    }
    // prefetch next K-tile of A (gfx1250 global_prefetch_b8)
    if (k0 + GM_BK < K)
      __builtin_prefetch(A + (long)(m0 + (tid >> 1)) * lda + k0 + GM_BK, 0, 1);
    wait_async0();
    __syncthreads();

    bf16x16 af[2], bfr[4];
#pragma unroll
    for (int i = 0; i < 2; ++i)
      af[i] = load_frag_rm(&As[32 * wr + 16 * i][0], GM_BK);
#pragma unroll
    for (int jb = 0; jb < 4; ++jb)
      bfr[jb] = load_frag_rm(&Bs[64 * wc + 16 * jb][0], GM_BK);
#pragma unroll
    for (int i = 0; i < 2; ++i)
#pragma unroll
      for (int jb = 0; jb < 4; ++jb)
        acc[i][jb] = wmma_bf16(af[i], bfr[jb], acc[i][jb]);
    __syncthreads();
  }

#pragma unroll
  for (int i = 0; i < 2; ++i)
#pragma unroll
    for (int jb = 0; jb < 4; ++jb)
#pragma unroll
      for (int j = 0; j < 8; ++j) {
        int row = m0 + 32 * wr + 16 * i + j + 8 * half;
        int col = n0 + 64 * wc + 16 * jb + l15;
        if (row < M && col < N) {
          float v = acc[i][jb][j] * alpha;
          if (c_bf16)
            ((__bf16*)Cv)[coff + (long)row * ldc + col] = (__bf16)v;
          else
            ((float*)Cv)[coff + (long)row * ldc + col] = v;
        }
      }
}

// ---------------- RMSNorm (f32 in -> bf16 out) ----------------
__global__ __launch_bounds__(256)
void rmsnorm_bf16(const float* __restrict__ in, const float* __restrict__ g,
                  __bf16* __restrict__ out, int N) {
  const long row = blockIdx.x;
  in  += row * (long)N;
  out += row * (long)N;
  __shared__ float red[256];
  float ss = 0.0f;
  for (int c = threadIdx.x; c < N; c += 256) {
    float v = in[c];
    ss += v * v;
  }
  red[threadIdx.x] = ss;
  __syncthreads();
  for (int s = 128; s > 0; s >>= 1) {
    if (threadIdx.x < s) red[threadIdx.x] += red[threadIdx.x + s];
    __syncthreads();
  }
  float r = rsqrtf(red[0] / (float)N + 1e-6f);
  for (int c = threadIdx.x; c < N; c += 256)
    out[c] = (__bf16)(in[c] * r * g[c]);
}

// ---------------- RoPE ----------------
__global__ __launch_bounds__(256)
void rope_q_kernel(const float* __restrict__ qr, const float* __restrict__ cosT,
                   const float* __restrict__ sinT, __bf16* __restrict__ out) {
  const int s = blockIdx.x;
  for (int p = threadIdx.x; p < NH * (DROT / 2); p += 256) {
    int h = p >> 5;
    int i = p & 31;
    float c  = cosT[(long)s * 32 + i];
    float sn = sinT[(long)s * 32 + i];
    long  ib = (long)s * (NH * DROT) + h * DROT + 2 * i;
    float x0 = qr[ib], x1 = qr[ib + 1];
    long  ob = ((long)h * SEQ + s) * DROT + 2 * i;
    out[ob]     = (__bf16)(x0 * c - x1 * sn);
    out[ob + 1] = (__bf16)(x0 * sn + x1 * c);
  }
}

__global__ __launch_bounds__(256)
void rope_k_kernel(const float* __restrict__ kr, const float* __restrict__ cosT,
                   const float* __restrict__ sinT, __bf16* __restrict__ out) {
  const int s = blockIdx.x;
  for (int i = threadIdx.x; i < DROT / 2; i += 256) {
    float c  = cosT[(long)s * 32 + i];
    float sn = sinT[(long)s * 32 + i];
    long  ib = (long)s * DROT + 2 * i;
    float x0 = kr[ib], x1 = kr[ib + 1];
    out[ib]     = (__bf16)(x0 * c - x1 * sn);
    out[ib + 1] = (__bf16)(x0 * sn + x1 * c);
  }
}

// ---------------- flash attention over compressed cache ----------------
// grid: (SEQ/64, NH); block 256 (8 waves). Q block = 64 rows, KV tile = 32.
__global__ __launch_bounds__(256)
void mla_flash_attn(const __bf16* __restrict__ qc, const __bf16* __restrict__ qrr,
                    const __bf16* __restrict__ ckv, const __bf16* __restrict__ krr,
                    __bf16* __restrict__ ocat, float scale) {
  __shared__ __bf16 kv_s[32][DKV + 8];   // 33,280 B  [k][c]  (TDM-padded rows)
  __shared__ __bf16 kr_s[32][DROT + 8];  //  4,608 B  [k][d]
  __shared__ float  s_s[64][33];
  __shared__ __bf16 p_s[64][40];
  __shared__ float  m_s[64], l_s[64], a_s[64], inv_s[64];

  const int h  = blockIdx.y;
  const int q0 = blockIdx.x * 64;
  const int tid  = threadIdx.x;
  const int wave = tid >> 5;
  const int lane = tid & 31;
  const int half = lane >> 4;
  const int l15  = lane & 15;
  const int rb_s = wave >> 1;  // S-phase rowblock 0..3
  const int cb_s = wave & 1;   // S-phase colblock 0..1

  // Q fragments held in registers for the whole block
  bf16x16 aq[DKV / 32];  // 16 frags
  bf16x16 ar[DROT / 32]; // 2 frags
  {
    const __bf16* qrow = qc + ((long)h * SEQ + q0 + rb_s * 16) * DKV;
#pragma unroll
    for (int kt = 0; kt < DKV / 32; ++kt)
      aq[kt] = load_frag_rm(qrow + kt * 32, DKV);
    const __bf16* qrrow = qrr + ((long)h * SEQ + q0 + rb_s * 16) * DROT;
#pragma unroll
    for (int kt = 0; kt < DROT / 32; ++kt)
      ar[kt] = load_frag_rm(qrrow + kt * 32, DROT);
  }

  floatx8 o_acc[4][4];   // rows 0..63 x cols [64*wave .. 64*wave+63]
#pragma unroll
  for (int i = 0; i < 4; ++i)
#pragma unroll
    for (int j = 0; j < 4; ++j) o_acc[i][j] = zero8();

  if (tid < 64) { m_s[tid] = -1e30f; l_s[tid] = 0.0f; }
  __syncthreads();

  const unsigned kv_lds = lds_addr_of(&kv_s[0][0]);
  const int kend = q0 + 64;  // causal upper bound (exclusive)
  for (int k0 = 0; k0 < kend; k0 += 32) {
    // KV tile via Tensor Data Mover (wave 0 issues, TENSORcnt drains)
    if (tid < 32) {
      tdm_load_kv_tile(kv_lds, ckv + (long)k0 * DKV, 32);
      __builtin_amdgcn_s_wait_tensorcnt(0);
    }
    // k_r tile via async global->LDS
    {
      int r = tid >> 3, cb = (tid & 7) * 8;
      async_load_b128(lds_addr_of(&kr_s[r][cb]),
                      &krr[(long)(k0 + r) * DROT + cb]);
      wait_async0();
    }
    __syncthreads();

    // scores: S[64 x 32] = q_c . c_kv^T + q_r . k_r^T
    floatx8 sfrag = zero8();
#pragma unroll
    for (int kt = 0; kt < DKV / 32; ++kt) {
      bf16x16 bfr = load_frag_rm(&kv_s[cb_s * 16][kt * 32], DKV + 8);
      sfrag = wmma_bf16(aq[kt], bfr, sfrag);
    }
#pragma unroll
    for (int kt = 0; kt < DROT / 32; ++kt) {
      bf16x16 bfr = load_frag_rm(&kr_s[cb_s * 16][kt * 32], DROT + 8);
      sfrag = wmma_bf16(ar[kt], bfr, sfrag);
    }
    // write scaled + causally-masked scores to LDS
#pragma unroll
    for (int j = 0; j < 8; ++j) {
      int rq = rb_s * 16 + j + 8 * half;  // 0..63
      int ck = cb_s * 16 + l15;           // 0..31
      float v = sfrag[j] * scale;
      if (k0 + ck > q0 + rq) v = -1e30f;
      s_s[rq][ck] = v;
    }
    __syncthreads();

    // online softmax: thread t owns row t
    if (tid < 64) {
      float m_old = m_s[tid];
      float mx = m_old;
      for (int c = 0; c < 32; ++c) mx = fmaxf(mx, s_s[tid][c]);
      float alpha = __expf(m_old - mx);
      float sum = 0.0f;
      for (int c = 0; c < 32; ++c) {
        float p = __expf(s_s[tid][c] - mx);
        p_s[tid][c] = (__bf16)p;
        sum += p;
      }
      m_s[tid] = mx;
      l_s[tid] = l_s[tid] * alpha + sum;
      a_s[tid] = alpha;
    }
    __syncthreads();

    // O = O*alpha + P @ V   (each wave owns 64 cols of DKV)
#pragma unroll
    for (int rb = 0; rb < 4; ++rb) {
      bf16x16 pa = load_frag_rm(&p_s[rb * 16][0], 40);
#pragma unroll
      for (int cb = 0; cb < 4; ++cb) {
#pragma unroll
        for (int j = 0; j < 8; ++j) {
          int rq = rb * 16 + j + 8 * half;
          o_acc[rb][cb][j] *= a_s[rq];
        }
        bf16x16 vb = load_frag_km(&kv_s[0][64 * wave + cb * 16], DKV + 8);
        o_acc[rb][cb] = wmma_bf16(pa, vb, o_acc[rb][cb]);
      }
    }
    __syncthreads();
  }

  if (tid < 64) inv_s[tid] = 1.0f / l_s[tid];
  __syncthreads();

#pragma unroll
  for (int rb = 0; rb < 4; ++rb)
#pragma unroll
    for (int cb = 0; cb < 4; ++cb)
#pragma unroll
      for (int j = 0; j < 8; ++j) {
        int rq  = rb * 16 + j + 8 * half;
        int col = 64 * wave + cb * 16 + l15;
        ocat[(long)(q0 + rq) * HC + h * DKV + col] =
            (__bf16)(o_acc[rb][cb][j] * inv_s[rq]);
      }
}

// ---------------- host-side orchestration ----------------
static inline dim3 gemm_grid(int M, int N, int batch) {
  return dim3((N + GM_BN - 1) / GM_BN, (M + GM_BM - 1) / GM_BM, batch);
}

extern "C" void kernel_launch(void* const* d_in, const int* in_sizes, int n_in,
                              void* d_out, int out_size, void* d_ws, size_t ws_size,
                              hipStream_t stream) {
  (void)in_sizes; (void)n_in; (void)out_size; (void)ws_size;

  const float* x     = (const float*)d_in[0];
  const float* w_dq  = (const float*)d_in[1];
  const float* w_uq  = (const float*)d_in[2];
  const float* w_dkv = (const float*)d_in[3];
  const float* w_uk  = (const float*)d_in[4];
  const float* w_uv  = (const float*)d_in[5];
  const float* w_qr  = (const float*)d_in[6];
  const float* w_kr  = (const float*)d_in[7];
  const float* w_o   = (const float*)d_in[8];
  const float* g_q   = (const float*)d_in[9];
  const float* g_kv  = (const float*)d_in[10];
  const float* fcos  = (const float*)d_in[11];
  const float* fsin  = (const float*)d_in[12];
  float* out = (float*)d_out;

  char* ws = (char*)d_ws;
  size_t off = 0;
  auto carve = [&](size_t bytes) -> char* {
    char* p = ws + off;
    off = (off + bytes + 255) & ~(size_t)255;
    return p;
  };
  __bf16* xb      = (__bf16*)carve((size_t)SEQ * DMODEL * 2);
  __bf16* wdq_b   = (__bf16*)carve((size_t)DQC * DMODEL * 2);
  __bf16* wdkv_b  = (__bf16*)carve((size_t)DKV * DMODEL * 2);
  __bf16* wqr_b   = (__bf16*)carve((size_t)NH * DROT * DQC * 2);
  __bf16* wkr_b   = (__bf16*)carve((size_t)DROT * DMODEL * 2);
  __bf16* wo_b    = (__bf16*)carve((size_t)DMODEL * NH * DHEAD * 2);
  __bf16* wuqT_b  = (__bf16*)carve((size_t)NH * DQC * DHEAD * 2);
  __bf16* wukT_b  = (__bf16*)carve((size_t)NH * DKV * DHEAD * 2);
  __bf16* wuvT_b  = (__bf16*)carve((size_t)NH * DKV * DHEAD * 2);
  float*  cq_f    = (float*)carve((size_t)SEQ * DQC * 4);
  float*  ckv_f   = (float*)carve((size_t)SEQ * DKV * 4);
  __bf16* cq_b    = (__bf16*)carve((size_t)SEQ * DQC * 2);
  __bf16* ckv_b   = (__bf16*)carve((size_t)SEQ * DKV * 2);
  float*  qr_f    = (float*)carve((size_t)SEQ * NH * DROT * 4);
  float*  kr_f    = (float*)carve((size_t)SEQ * DROT * 4);
  __bf16* qr_b    = (__bf16*)carve((size_t)NH * SEQ * DROT * 2);
  __bf16* kr_b    = (__bf16*)carve((size_t)SEQ * DROT * 2);
  __bf16* wukuq_b = (__bf16*)carve((size_t)NH * DKV * DQC * 2);
  __bf16* qc_b    = (__bf16*)carve((size_t)NH * SEQ * DKV * 2);
  __bf16* wuov_b  = (__bf16*)carve((size_t)DMODEL * HC * 2);
  __bf16* ocat_b  = (__bf16*)carve((size_t)SEQ * HC * 2);

  const dim3 blk(256);

  // --- convert / transpose weights and activations to bf16 ---
  conv_bf16_3d<<<SEQ, blk, 0, stream>>>(x, xb, 1, DMODEL, DMODEL, 0, 1);
  conv_bf16_3d<<<DQC, blk, 0, stream>>>(w_dq, wdq_b, 1, DMODEL, DMODEL, 0, 1);
  conv_bf16_3d<<<DKV, blk, 0, stream>>>(w_dkv, wdkv_b, 1, DMODEL, DMODEL, 0, 1);
  conv_bf16_3d<<<NH * DROT, blk, 0, stream>>>(w_qr, wqr_b, 1, DQC, DQC, 0, 1);
  conv_bf16_3d<<<DROT, blk, 0, stream>>>(w_kr, wkr_b, 1, DMODEL, DMODEL, 0, 1);
  conv_bf16_3d<<<DMODEL, blk, 0, stream>>>(w_o, wo_b, 1, NH * DHEAD, NH * DHEAD, 0, 1);
  conv_bf16_3d<<<NH * DQC, blk, 0, stream>>>(w_uq, wuqT_b, DQC, DHEAD,
                                             (long)DHEAD * DQC, 1, DQC);
  conv_bf16_3d<<<NH * DKV, blk, 0, stream>>>(w_uk, wukT_b, DKV, DHEAD,
                                             (long)DHEAD * DKV, 1, DKV);
  conv_bf16_3d<<<NH * DKV, blk, 0, stream>>>(w_uv, wuvT_b, DKV, DHEAD,
                                             (long)DHEAD * DKV, 1, DKV);

  // --- down projections + RMSNorm ---
  gemm_bf16_wmma<<<gemm_grid(SEQ, DQC, 1), blk, 0, stream>>>(
      xb, 0, DMODEL, wdq_b, 0, DMODEL, cq_f, 0, DQC, SEQ, DQC, DMODEL, 0, 1.0f);
  gemm_bf16_wmma<<<gemm_grid(SEQ, DKV, 1), blk, 0, stream>>>(
      xb, 0, DMODEL, wdkv_b, 0, DMODEL, ckv_f, 0, DKV, SEQ, DKV, DMODEL, 0, 1.0f);
  rmsnorm_bf16<<<SEQ, blk, 0, stream>>>(cq_f, g_q, cq_b, DQC);
  rmsnorm_bf16<<<SEQ, blk, 0, stream>>>(ckv_f, g_kv, ckv_b, DKV);

  // --- decoupled RoPE branch ---
  gemm_bf16_wmma<<<gemm_grid(SEQ, NH * DROT, 1), blk, 0, stream>>>(
      cq_b, 0, DQC, wqr_b, 0, DQC, qr_f, 0, NH * DROT, SEQ, NH * DROT, DQC, 0, 1.0f);
  gemm_bf16_wmma<<<gemm_grid(SEQ, DROT, 1), blk, 0, stream>>>(
      xb, 0, DMODEL, wkr_b, 0, DMODEL, kr_f, 0, DROT, SEQ, DROT, DMODEL, 0, 1.0f);
  rope_q_kernel<<<SEQ, blk, 0, stream>>>(qr_f, fcos, fsin, qr_b);
  rope_k_kernel<<<SEQ, blk, 0, stream>>>(kr_f, fcos, fsin, kr_b);

  // --- absorbed weights: wukuq[h][c][q] = sum_d w_uk[d,c] * w_uq[d,q] ---
  gemm_bf16_wmma<<<gemm_grid(DKV, DQC, NH), blk, 0, stream>>>(
      wukT_b, (long)DKV * DHEAD, DHEAD,
      wuqT_b, (long)DQC * DHEAD, DHEAD,
      wukuq_b, (long)DKV * DQC, DQC, DKV, DQC, DHEAD, 1, 1.0f);

  // --- q_c[h][s][c] = sum_q cq[s,q] * wukuq[h][c][q] ---
  gemm_bf16_wmma<<<gemm_grid(SEQ, DKV, NH), blk, 0, stream>>>(
      cq_b, 0, DQC,
      wukuq_b, (long)DKV * DQC, DQC,
      qc_b, (long)SEQ * DKV, DKV, SEQ, DKV, DQC, 1, 1.0f);

  // --- absorbed output weights: wuov[m][h*512+c] = sum_d w_o[m,h*DH+d]*w_uv[d,c] ---
  gemm_bf16_wmma<<<gemm_grid(DMODEL, DKV, NH), blk, 0, stream>>>(
      wo_b, (long)DHEAD, NH * DHEAD,
      wuvT_b, (long)DKV * DHEAD, DHEAD,
      wuov_b, (long)DKV, HC, DMODEL, DKV, DHEAD, 1, 1.0f);

  // --- flash attention over compressed cache ---
  const float scale = 1.0f / sqrtf((float)(DHEAD + DROT));
  mla_flash_attn<<<dim3(SEQ / 64, NH), blk, 0, stream>>>(
      qc_b, qr_b, ckv_b, kr_b, ocat_b, scale);

  // --- final projection: out[s,m] = sum_{hc} ocat[s,hc] * wuov[m,hc] ---
  gemm_bf16_wmma<<<gemm_grid(SEQ, DMODEL, 1), blk, 0, stream>>>(
      ocat_b, 0, HC, wuov_b, 0, HC, out, 0, DMODEL, SEQ, DMODEL, HC, 0, 1.0f);
}
